// NeuroSAT_48885317763226
// MI455X (gfx1250) — matrix operand order repaired and checked by reading, not compiled, not run on hardware
//
#include <hip/hip_runtime.h>
#include <cstdint>
#include <cstddef>

#define DIM 128
#define NVARS 4096
#define NLITS 8192
#define NCLS 16384
#define NCELLS 98304
#define NPROBS 32
#define NROUNDS 16

typedef __bf16 bf16;
typedef __attribute__((ext_vector_type(8)))  __bf16 v8bf;
typedef __attribute__((ext_vector_type(16))) __bf16 v16bf;
typedef __attribute__((ext_vector_type(8)))  float  v8f;

__device__ __forceinline__ bf16 f2bf(float f) {
  unsigned u = __float_as_uint(f);
  unsigned short h = (unsigned short)((u + 0x7FFFu + ((u >> 16) & 1u)) >> 16); // RNE
  return __builtin_bit_cast(bf16, h);
}
__device__ __forceinline__ float sigf(float x) { return 1.f / (1.f + expf(-x)); }

// ---------------- utility kernels ----------------
__global__ void k_zero_f32(float* p, int n) {
  int i = blockIdx.x * blockDim.x + threadIdx.x;
  if (i < n) p[i] = 0.f;
}
__global__ void k_zero_u32(unsigned* p, int n) {
  int i = blockIdx.x * blockDim.x + threadIdx.x;
  if (i < n) p[i] = 0u;
}
__global__ void k_f32_to_bf16(const float* __restrict__ s, bf16* __restrict__ d, int n) {
  int i = blockIdx.x * blockDim.x + threadIdx.x;
  if (i < n) d[i] = f2bf(s[i]);
}
__global__ void k_add_vec(const float* __restrict__ a, const float* __restrict__ b,
                          float* __restrict__ o, int n) {
  int i = blockIdx.x * blockDim.x + threadIdx.x;
  if (i < n) o[i] = a[i] + b[i];
}
__global__ void k_init_state(float* __restrict__ h, float* __restrict__ c,
                             const float* __restrict__ w, const float* __restrict__ b, int n) {
  int i = blockIdx.x * blockDim.x + threadIdx.x;
  if (i < n) { int d = i & (DIM - 1); h[i] = w[d] + b[d]; c[i] = 0.f; }
}

// ---------------- adjacency setup ----------------
__global__ void k_dedupe(const int* __restrict__ lit, const int* __restrict__ cla,
                         unsigned* __restrict__ bits, int* __restrict__ rd,
                         int* __restrict__ cd, float* __restrict__ cw) {
  int e = blockIdx.x * blockDim.x + threadIdx.x;
  if (e >= NCELLS) return;
  int l = lit[e], c = cla[e];
  unsigned long long bi = (unsigned long long)l * NCLS + (unsigned)c;
  unsigned m = 1u << (bi & 31);
  unsigned old = atomicOr(&bits[bi >> 5], m);
  int valid = (old & m) == 0;
  cw[e] = valid ? 1.f : 0.f;
  if (valid) { atomicAdd(&rd[l], 1); atomicAdd(&cd[c], 1); }
}
__global__ void k_norm_w(const int* __restrict__ lit, const int* __restrict__ cla,
                         const int* __restrict__ rd, const int* __restrict__ cd,
                         float* __restrict__ cw) {
  int e = blockIdx.x * blockDim.x + threadIdx.x;
  if (e >= NCELLS) return;
  if (cw[e] != 0.f) {
    float rv = 1.f / (sqrtf((float)rd[lit[e]]) + 1e-6f);
    float cv = 1.f / (sqrtf((float)cd[cla[e]]) + 1e-6f);
    cw[e] = rv * cv;
  }
}

// SpMM via scatter atomics: one block (128 threads) per nonzero cell.
__global__ void k_scatter(const int* __restrict__ lit, const int* __restrict__ cla,
                          const float* __restrict__ w, const float* __restrict__ src,
                          float* __restrict__ dst, int toClause) {
  int e = blockIdx.x;
  int d = threadIdx.x;
  float wv = w[e];
  if (wv == 0.f) return;          // uniform per block
  int l = lit[e], c = cla[e];
  int si = toClause ? l : c;
  int di = toClause ? c : l;
  float v = wv * src[(size_t)si * DIM + d];
  __hip_atomic_fetch_add(&dst[(size_t)di * DIM + d], v,
                         __ATOMIC_RELAXED, __HIP_MEMORY_SCOPE_AGENT);
}

// ---------------- WMMA GEMM: Y[M,N] = act(X[M,K] @ W[N,K]^T + bias) ----------------
// flags bit0 = ReLU, bit1 = accumulate into existing Y (bias ignored).
// Register-blocked: each wave computes a 32x64 macro-tile = 2 A-frags x 4 B-frags
// -> 8 WMMAs per K-step for 12 b128 loads (A reused 4x, B reused 2x; B = weights,
// hot in L2/WGP$ across the grid). Block = 8 waves (4 along M, 2 along N) covers
// 128x128 of output. Requires M%128==0, N%128==0, K%32==0 (true for all calls).
__global__ __launch_bounds__(256) void k_gemm_bf16(
    const bf16* __restrict__ X, const bf16* __restrict__ W,
    const float* __restrict__ bias, float* __restrict__ Y,
    int M, int N, int K, int flags) {
  const int lane = threadIdx.x & 31;
  const int wv = threadIdx.x >> 5;
  const int mw = wv & 3;
  const int nw = wv >> 2;
  const int m0 = blockIdx.y * 128 + mw * 32;
  const int n0 = blockIdx.x * 128 + nw * 64;
  const int lr = lane & 15;
  const int hi = lane >> 4;                  // 0|1: K-half select
  // A frag: lane = row, one contiguous 16-elt bf16 run per K-step (two 8-elt loads)
  const bf16* xp0 = X + (size_t)(m0 + lr) * K + hi * 16;
  const bf16* xp1 = xp0 + (size_t)16 * K;
  // B frag: lane = output col = W row, one contiguous 16-elt run per K-step
  const bf16* wp = W + (size_t)(n0 + lr) * K + hi * 16;
  const size_t ws = (size_t)16 * K;

  v8f acc[2][4];
#pragma unroll
  for (int i = 0; i < 2; ++i)
#pragma unroll
    for (int j = 0; j < 4; ++j)
      acc[i][j] = (v8f){0.f, 0.f, 0.f, 0.f, 0.f, 0.f, 0.f, 0.f};

  for (int kk = 0; kk < K; kk += 32) {
    v8bf a0l = *reinterpret_cast<const v8bf*>(xp0 + kk);
    v8bf a0h = *reinterpret_cast<const v8bf*>(xp0 + kk + 8);
    v8bf a1l = *reinterpret_cast<const v8bf*>(xp1 + kk);
    v8bf a1h = *reinterpret_cast<const v8bf*>(xp1 + kk + 8);
    v16bf a0 = __builtin_shufflevector(a0l, a0h, 0, 1, 2, 3, 4, 5, 6, 7,
                                       8, 9, 10, 11, 12, 13, 14, 15);
    v16bf a1 = __builtin_shufflevector(a1l, a1h, 0, 1, 2, 3, 4, 5, 6, 7,
                                       8, 9, 10, 11, 12, 13, 14, 15);
    v16bf b0 = *reinterpret_cast<const v16bf*>(wp + kk);
    v16bf b1 = *reinterpret_cast<const v16bf*>(wp + ws + kk);
    v16bf b2 = *reinterpret_cast<const v16bf*>(wp + 2 * ws + kk);
    v16bf b3 = *reinterpret_cast<const v16bf*>(wp + 3 * ws + kk);
    acc[0][0] = __builtin_amdgcn_wmma_f32_16x16x32_bf16(false, a0, false, b0, (short)0, acc[0][0], false, false);
    acc[0][1] = __builtin_amdgcn_wmma_f32_16x16x32_bf16(false, a0, false, b1, (short)0, acc[0][1], false, false);
    acc[0][2] = __builtin_amdgcn_wmma_f32_16x16x32_bf16(false, a0, false, b2, (short)0, acc[0][2], false, false);
    acc[0][3] = __builtin_amdgcn_wmma_f32_16x16x32_bf16(false, a0, false, b3, (short)0, acc[0][3], false, false);
    acc[1][0] = __builtin_amdgcn_wmma_f32_16x16x32_bf16(false, a1, false, b0, (short)0, acc[1][0], false, false);
    acc[1][1] = __builtin_amdgcn_wmma_f32_16x16x32_bf16(false, a1, false, b1, (short)0, acc[1][1], false, false);
    acc[1][2] = __builtin_amdgcn_wmma_f32_16x16x32_bf16(false, a1, false, b2, (short)0, acc[1][2], false, false);
    acc[1][3] = __builtin_amdgcn_wmma_f32_16x16x32_bf16(false, a1, false, b3, (short)0, acc[1][3], false, false);
  }

  // D layout: lanes 0-15 -> rows r+0, lanes 16-31 -> rows r+8; col = lr.
#pragma unroll
  for (int i = 0; i < 2; ++i) {
    const int mb = m0 + i * 16 + hi * 8;
#pragma unroll
    for (int j = 0; j < 4; ++j) {
      const int n = n0 + j * 16 + lr;
      float bv = (bias && !(flags & 2)) ? bias[n] : 0.f;
#pragma unroll
      for (int r = 0; r < 8; ++r) {
        size_t off = (size_t)(mb + r) * N + n;
        float v = acc[i][j][r] + ((flags & 2) ? Y[off] : bv);
        if (flags & 1) v = v > 0.f ? v : 0.f;
        Y[off] = v;
      }
    }
  }
}

// ---------------- LSTM pointwise (pytorch gate order i,f,g,o) ----------------
__global__ void k_lstm(const float* __restrict__ G, float* __restrict__ h,
                       float* __restrict__ c, int n) {
  int i = blockIdx.x * blockDim.x + threadIdx.x;
  if (i >= n) return;
  int r = i >> 7, d = i & 127;
  const float* g = G + (size_t)r * 4 * DIM;
  float ig = sigf(g[d]);
  float fg = sigf(g[DIM + d]);
  float gg = tanhf(g[2 * DIM + d]);
  float og = sigf(g[3 * DIM + d]);
  float c2 = fg * c[i] + ig * gg;
  c[i] = c2;
  h[i] = og * tanhf(c2);
}

// x = [CL, flip(L_h)] as bf16, row stride 2*DIM
__global__ void k_concat_flip(const float* __restrict__ CL, const float* __restrict__ Lh,
                              bf16* __restrict__ x) {
  int i = blockIdx.x * blockDim.x + threadIdx.x;
  if (i >= NLITS * DIM) return;
  int l = i >> 7, d = i & 127;
  x[(size_t)l * 2 * DIM + d] = f2bf(CL[i]);
  int lf = (l < NVARS) ? l + NVARS : l - NVARS;
  x[(size_t)l * 2 * DIM + DIM + d] = f2bf(Lh[(size_t)lf * DIM + d]);
}

// last vote layer: [8192,128] x [128] + b
__global__ void k_vote(const float* __restrict__ t2, const float* __restrict__ w3,
                       const float* __restrict__ b3, float* __restrict__ vote) {
  int r = blockIdx.x * blockDim.x + threadIdx.x;
  if (r >= NLITS) return;
  float s = b3[0];
  const float* row = t2 + (size_t)r * DIM;
#pragma unroll 8
  for (int k = 0; k < DIM; ++k) s += row[k] * w3[k];
  vote[r] = s;
}

__global__ void k_vote_mean(const float* __restrict__ vote, float* __restrict__ out) {
  __shared__ float s[256];
  int p = blockIdx.x, q = threadIdx.x;
  int f = p * 256 + q;
  int v = f >> 1, j = f & 1;                 // vote_join[v,j] flattened
  s[q] = vote[v + j * NVARS];
  __syncthreads();
  for (int st = 128; st > 0; st >>= 1) {
    if (q < st) s[q] += s[q + st];
    __syncthreads();
  }
  if (q == 0) out[p] = s[0] * (1.f / 256.f);
}

// ---------------- host orchestration ----------------
extern "C" void kernel_launch(void* const* d_in, const int* in_sizes, int n_in,
                              void* d_out, int out_size, void* d_ws, size_t ws_size,
                              hipStream_t stream) {
  const int* lit = (const int*)d_in[0];
  const int* cla = (const int*)d_in[1];
  const float* L_init_w = (const float*)d_in[2];
  const float* L_init_b = (const float*)d_in[3];
  const float* C_init_w = (const float*)d_in[4];
  const float* C_init_b = (const float*)d_in[5];
  const float* lm_w[3] = {(const float*)d_in[6], (const float*)d_in[8], (const float*)d_in[10]};
  const float* lm_b[3] = {(const float*)d_in[7], (const float*)d_in[9], (const float*)d_in[11]};
  const float* cm_w[3] = {(const float*)d_in[12], (const float*)d_in[14], (const float*)d_in[16]};
  const float* cm_b[3] = {(const float*)d_in[13], (const float*)d_in[15], (const float*)d_in[17]};
  const float* cu_wih = (const float*)d_in[18];
  const float* cu_whh = (const float*)d_in[19];
  const float* cu_bih = (const float*)d_in[20];
  const float* cu_bhh = (const float*)d_in[21];
  const float* lu_wih = (const float*)d_in[22];
  const float* lu_whh = (const float*)d_in[23];
  const float* lu_bih = (const float*)d_in[24];
  const float* lu_bhh = (const float*)d_in[25];
  const float* lv_w1 = (const float*)d_in[26];
  const float* lv_b1 = (const float*)d_in[27];
  const float* lv_w2 = (const float*)d_in[28];
  const float* lv_b2 = (const float*)d_in[29];
  const float* lv_w3 = (const float*)d_in[30];
  const float* lv_b3 = (const float*)d_in[31];
  float* out = (float*)d_out;

  uint8_t* base = (uint8_t*)d_ws;
  size_t off = 0;
  auto alloc = [&](size_t bytes) -> void* {
    void* r = base + off;
    off += (bytes + 255) & ~(size_t)255;
    return r;
  };
  // fp32 state / scratch
  unsigned* bits = (unsigned*)alloc((size_t)NLITS * NCLS / 8);
  int* rowdeg = (int*)alloc(NLITS * 4);
  int* coldeg = (int*)alloc(NCLS * 4);
  float* cellw = (float*)alloc(NCELLS * 4);
  float* L_h = (float*)alloc((size_t)NLITS * DIM * 4);
  float* L_c = (float*)alloc((size_t)NLITS * DIM * 4);
  float* C_h = (float*)alloc((size_t)NCLS * DIM * 4);
  float* C_c = (float*)alloc((size_t)NCLS * DIM * 4);
  float* L_pre = (float*)alloc((size_t)NLITS * DIM * 4);
  float* C_pre = (float*)alloc((size_t)NCLS * DIM * 4);
  float* LC = (float*)alloc((size_t)NCLS * DIM * 4);
  float* CL = (float*)alloc((size_t)NLITS * DIM * 4);
  float* tmpL = (float*)alloc((size_t)NLITS * DIM * 4);
  float* tmpC = (float*)alloc((size_t)NCLS * DIM * 4);
  float* G_c = (float*)alloc((size_t)NCLS * 4 * DIM * 4);
  float* G_l = (float*)alloc((size_t)NLITS * 4 * DIM * 4);
  float* vote = (float*)alloc(NLITS * 4);
  float* cu_bias = (float*)alloc(4 * DIM * 4);
  float* lu_bias = (float*)alloc(4 * DIM * 4);
  // bf16 weights
  bf16* lm_wb[3]; bf16* cm_wb[3];
  for (int i = 0; i < 3; ++i) lm_wb[i] = (bf16*)alloc(DIM * DIM * 2);
  for (int i = 0; i < 3; ++i) cm_wb[i] = (bf16*)alloc(DIM * DIM * 2);
  bf16* lv_w1b = (bf16*)alloc(DIM * DIM * 2);
  bf16* lv_w2b = (bf16*)alloc(DIM * DIM * 2);
  bf16* cu_wihb = (bf16*)alloc(4 * DIM * DIM * 2);
  bf16* cu_whhb = (bf16*)alloc(4 * DIM * DIM * 2);
  bf16* lu_wihb = (bf16*)alloc(4 * DIM * 2 * DIM * 2);
  bf16* lu_whhb = (bf16*)alloc(4 * DIM * DIM * 2);
  // bf16 activation staging
  bf16* bfL = (bf16*)alloc((size_t)NLITS * DIM * 2);
  bf16* bfC = (bf16*)alloc((size_t)NCLS * DIM * 2);
  bf16* xcat = (bf16*)alloc((size_t)NLITS * 2 * DIM * 2);
  (void)ws_size; (void)in_sizes; (void)n_in; (void)out_size;

  auto conv = [&](const float* s, bf16* d, int n) {
    k_f32_to_bf16<<<(n + 255) / 256, 256, 0, stream>>>(s, d, n);
  };
  auto zerof = [&](float* p, int n) {
    k_zero_f32<<<(n + 255) / 256, 256, 0, stream>>>(p, n);
  };
  auto gemm = [&](const bf16* X, const bf16* W, const float* bias, float* Y,
                  int M, int N, int K, int flags) {
    dim3 g(N / 128, M / 128);
    k_gemm_bf16<<<g, 256, 0, stream>>>(X, W, bias, Y, M, N, K, flags);
  };

  // ---- one-time per call: weight conversion + adjacency normalization ----
  for (int i = 0; i < 3; ++i) conv(lm_w[i], lm_wb[i], DIM * DIM);
  for (int i = 0; i < 3; ++i) conv(cm_w[i], cm_wb[i], DIM * DIM);
  conv(lv_w1, lv_w1b, DIM * DIM);
  conv(lv_w2, lv_w2b, DIM * DIM);
  conv(cu_wih, cu_wihb, 4 * DIM * DIM);
  conv(cu_whh, cu_whhb, 4 * DIM * DIM);
  conv(lu_wih, lu_wihb, 4 * DIM * 2 * DIM);
  conv(lu_whh, lu_whhb, 4 * DIM * DIM);
  k_add_vec<<<2, 256, 0, stream>>>(cu_bih, cu_bhh, cu_bias, 4 * DIM);
  k_add_vec<<<2, 256, 0, stream>>>(lu_bih, lu_bhh, lu_bias, 4 * DIM);

  int nbw = NLITS * NCLS / 32;
  k_zero_u32<<<(nbw + 255) / 256, 256, 0, stream>>>(bits, nbw);
  k_zero_u32<<<(NLITS + 255) / 256, 256, 0, stream>>>((unsigned*)rowdeg, NLITS);
  k_zero_u32<<<(NCLS + 255) / 256, 256, 0, stream>>>((unsigned*)coldeg, NCLS);
  k_dedupe<<<(NCELLS + 255) / 256, 256, 0, stream>>>(lit, cla, bits, rowdeg, coldeg, cellw);
  k_norm_w<<<(NCELLS + 255) / 256, 256, 0, stream>>>(lit, cla, rowdeg, coldeg, cellw);

  k_init_state<<<(NLITS * DIM + 255) / 256, 256, 0, stream>>>(L_h, L_c, L_init_w, L_init_b, NLITS * DIM);
  k_init_state<<<(NCLS * DIM + 255) / 256, 256, 0, stream>>>(C_h, C_c, C_init_w, C_init_b, NCLS * DIM);

  // ---- 16 message-passing rounds ----
  for (int r = 0; r < NROUNDS; ++r) {
    // L_pre = MLP3(L_h)
    conv(L_h, bfL, NLITS * DIM);
    gemm(bfL, lm_wb[0], lm_b[0], tmpL, NLITS, DIM, DIM, 1);
    conv(tmpL, bfL, NLITS * DIM);
    gemm(bfL, lm_wb[1], lm_b[1], tmpL, NLITS, DIM, DIM, 1);
    conv(tmpL, bfL, NLITS * DIM);
    gemm(bfL, lm_wb[2], lm_b[2], L_pre, NLITS, DIM, DIM, 0);
    // LC = A^T @ L_pre
    zerof(LC, NCLS * DIM);
    k_scatter<<<NCELLS, DIM, 0, stream>>>(lit, cla, cellw, L_pre, LC, 1);
    // C gates + LSTM
    conv(LC, bfC, NCLS * DIM);
    gemm(bfC, cu_wihb, cu_bias, G_c, NCLS, 4 * DIM, DIM, 0);
    conv(C_h, bfC, NCLS * DIM);
    gemm(bfC, cu_whhb, nullptr, G_c, NCLS, 4 * DIM, DIM, 2);
    k_lstm<<<(NCLS * DIM + 255) / 256, 256, 0, stream>>>(G_c, C_h, C_c, NCLS * DIM);
    // C_pre = MLP3(C_h)
    conv(C_h, bfC, NCLS * DIM);
    gemm(bfC, cm_wb[0], cm_b[0], tmpC, NCLS, DIM, DIM, 1);
    conv(tmpC, bfC, NCLS * DIM);
    gemm(bfC, cm_wb[1], cm_b[1], tmpC, NCLS, DIM, DIM, 1);
    conv(tmpC, bfC, NCLS * DIM);
    gemm(bfC, cm_wb[2], cm_b[2], C_pre, NCLS, DIM, DIM, 0);
    // CL = A @ C_pre
    zerof(CL, NLITS * DIM);
    k_scatter<<<NCELLS, DIM, 0, stream>>>(lit, cla, cellw, C_pre, CL, 0);
    // L gates + LSTM (x = [CL, flip(L_h)], h = old L_h)
    k_concat_flip<<<(NLITS * DIM + 255) / 256, 256, 0, stream>>>(CL, L_h, xcat);
    conv(L_h, bfL, NLITS * DIM);
    gemm(xcat, lu_wihb, lu_bias, G_l, NLITS, 4 * DIM, 2 * DIM, 0);
    gemm(bfL, lu_whhb, nullptr, G_l, NLITS, 4 * DIM, DIM, 2);
    k_lstm<<<(NLITS * DIM + 255) / 256, 256, 0, stream>>>(G_l, L_h, L_c, NLITS * DIM);
  }

  // ---- vote MLP + mean ----
  conv(L_h, bfL, NLITS * DIM);
  gemm(bfL, lv_w1b, lv_b1, tmpL, NLITS, DIM, DIM, 1);
  conv(tmpL, bfL, NLITS * DIM);
  gemm(bfL, lv_w2b, lv_b2, tmpL, NLITS, DIM, DIM, 1);
  k_vote<<<(NLITS + 255) / 256, 256, 0, stream>>>(tmpL, lv_w3, lv_b3, vote);
  k_vote_mean<<<NPROBS, 256, 0, stream>>>(vote, out);
}
